// TritonSoftmaxV_85624468013186
// MI455X (gfx1250) — compile-verified, matrix operand order checked by reading, not run on hardware
//
#include <hip/hip_runtime.h>

typedef __attribute__((ext_vector_type(16))) _Float16 v16h;
typedef __attribute__((ext_vector_type(8)))  _Float16 v8h;
typedef __attribute__((ext_vector_type(2)))  _Float16 v2h;
typedef __attribute__((ext_vector_type(8)))  float    v8f;
typedef __attribute__((ext_vector_type(4)))  float    v4f;

namespace {
constexpr int S_LEN = 2048;
constexpr int D_LEN = 128;
constexpr int KSTEP = 64;   // two WMMA k-steps per softmax update
constexpr int LDSS  = 72;   // transposed LDS row stride in halves (144 B: 16B-aligned, conflict-free)
constexpr float LOG2E = 1.44269504088896340736f;
}

__global__ __launch_bounds__(256)
void softmaxv_wmma_kernel(const float* __restrict__ scores,
                          const float* __restrict__ vsrc,
                          float* __restrict__ out)
{
    // Double-buffered V tile, staged transposed: vt[buf][d * LDSS + k], k in [0,64)
    __shared__ __align__(16) _Float16 vt[2][D_LEN * LDSS];   // 2 x 18,432 B

    const int tid  = threadIdx.x;
    const int lane = tid & 31;
    const int wave = tid >> 5;
    const int half = lane >> 4;     // which 16-lane half of the wave
    const int mrow = lane & 15;     // row (A) / column (B,C) index within 16-tile

    const int bh    = blockIdx.x >> 4;    // b*H + h  (0..31)
    const int qblk  = blockIdx.x & 15;    // q-block  (0..15)
    const int qbase = qblk * 128 + wave * 16;

    const float* srow = scores + ((size_t)bh * S_LEN + (qbase + mrow)) * S_LEN;
    const float* vbh  = vsrc   + (size_t)bh * S_LEN * D_LEN;

    // staging decomposition: 64x128 tile = 1024 chunks of (2k x 4d); 4 chunks/thread
    int sd0[4], skp[4];
#pragma unroll
    for (int c = 0; c < 4; ++c) {
        const int idx = tid + c * 256;
        sd0[c] = (idx & 31) << 2;   // d base 0..124
        skp[c] = idx >> 5;          // k-pair 0..31
    }

    v8f acc[8];
#pragma unroll
    for (int nt = 0; nt < 8; ++nt)
#pragma unroll
        for (int r = 0; r < 8; ++r) acc[nt][r] = 0.0f;

    float m_run = -__builtin_inff();   // running row max (base-2 domain)
    float s_run = 0.0f;                // running row sum (this lane's partial)

    // ---- prologue: stage tile 0 into vt[0] ----
#pragma unroll
    for (int c = 0; c < 4; ++c) {
        const int kk = 2 * skp[c];
        v4f xa = *(const v4f*)(vbh + (size_t)(kk)     * D_LEN + sd0[c]);
        v4f xb = *(const v4f*)(vbh + (size_t)(kk + 1) * D_LEN + sd0[c]);
#pragma unroll
        for (int j = 0; j < 4; ++j) {
            v2h p; p[0] = (_Float16)xa[j]; p[1] = (_Float16)xb[j];
            *(v2h*)&vt[0][(sd0[c] + j) * LDSS + kk] = p;
        }
    }
    __syncthreads();

    for (int k0 = 0; k0 < S_LEN; k0 += KSTEP) {
        const int  buf  = (k0 >> 6) & 1;
        const bool more = (k0 + KSTEP) < S_LEN;

        // ---- issue next V tile's global loads early (hidden behind softmax+WMMA) ----
        v4f nxa[4], nxb[4];
        if (more) {
            const int kn = k0 + KSTEP;
#pragma unroll
            for (int c = 0; c < 4; ++c) {
                const int kk = kn + 2 * skp[c];
                nxa[c] = *(const v4f*)(vbh + (size_t)(kk)     * D_LEN + sd0[c]);
                nxb[c] = *(const v4f*)(vbh + (size_t)(kk + 1) * D_LEN + sd0[c]);
            }
        }

        // ---- A operands: 32 fp32 scores per lane (row mrow; 2 k-steps of ISA layout) ----
        float sv[32];
#pragma unroll
        for (int h = 0; h < 2; ++h) {
            const int cb = k0 + h * 32 + half * 8;
            v4f sa = *(const v4f*)(srow + cb);
            v4f sb = *(const v4f*)(srow + cb + 4);
            v4f sc = *(const v4f*)(srow + cb + 16);
            v4f sd = *(const v4f*)(srow + cb + 20);
#pragma unroll
            for (int j = 0; j < 4; ++j) {
                sv[h * 16 + j]      = sa[j];
                sv[h * 16 + 4 + j]  = sb[j];
                sv[h * 16 + 8 + j]  = sc[j];
                sv[h * 16 + 12 + j] = sd[j];
            }
            if (more) __builtin_prefetch(srow + cb + KSTEP, 0, 1);
        }

        // ---- online softmax update (base-2 domain; max taken on raw scores) ----
        float mxs = sv[0];
#pragma unroll
        for (int j = 1; j < 32; ++j) mxs = fmaxf(mxs, sv[j]);
        mxs = fmaxf(mxs, __shfl_xor(mxs, 16, 32));     // combine lane pair (same row)
        const float m_new = fmaxf(m_run, mxs * LOG2E);
        const float scale = __builtin_amdgcn_exp2f(m_run - m_new);
        m_run = m_new;

        v16h a[2];
        float psum = 0.0f;
#pragma unroll
        for (int h = 0; h < 2; ++h)
#pragma unroll
            for (int j = 0; j < 16; ++j) {
                const float p = __builtin_amdgcn_exp2f(
                    fmaf(sv[h * 16 + j], LOG2E, -m_new));
                psum += p;
                a[h][j] = (_Float16)p;
            }
        s_run = s_run * scale + psum;

        // ---- store staged next tile into the other buffer (packed k-pairs) ----
        if (more) {
            _Float16* nb = vt[buf ^ 1];
#pragma unroll
            for (int c = 0; c < 4; ++c) {
                const int kk = 2 * skp[c];
#pragma unroll
                for (int j = 0; j < 4; ++j) {
                    v2h p; p[0] = (_Float16)nxa[c][j]; p[1] = (_Float16)nxb[c][j];
                    *(v2h*)&nb[(sd0[c] + j) * LDSS + kk] = p;
                }
            }
        }

        // rescale accumulators: scale is per softmax-row (lane-indexed); the C/D
        // layout maps VGPR r -> row r + 8*half, so remap via bpermute into a v8f.
        v8f svec;
#pragma unroll
        for (int r = 0; r < 8; ++r)
            svec[r] = __shfl(scale, r + (half << 3), 32);
#pragma unroll
        for (int nt = 0; nt < 8; ++nt) acc[nt] *= svec;

        // ---- B operands from transposed LDS (current buffer), 16 WMMAs ----
        const _Float16* vb = vt[buf];
#pragma unroll
        for (int h = 0; h < 2; ++h)
#pragma unroll
            for (int nt = 0; nt < 8; ++nt) {
                const _Float16* row =
                    &vb[(nt * 16 + mrow) * LDSS + h * 32 + half * 8];
                v8h blo = *(const v8h*)(row);        // K = base .. base+7
                v8h bhi = *(const v8h*)(row + 16);   // K = base+16 .. base+23
                v16h b = __builtin_shufflevector(blo, bhi,
                           0, 1, 2, 3, 4, 5, 6, 7, 8, 9, 10, 11, 12, 13, 14, 15);
                acc[nt] = __builtin_amdgcn_wmma_f32_16x16x32_f16(
                    false, a[h], false, b, (short)0, acc[nt], false, false);
            }

        __syncthreads();   // single barrier per 64-wide k-step (ping-pong safe)
    }

    // ---- finalize: divide by full row sums (remapped to C layout), store ----
    const float total = s_run + __shfl_xor(s_run, 16, 32);
    float* obase = out + ((size_t)bh * S_LEN + qbase) * D_LEN;
#pragma unroll
    for (int r = 0; r < 8; ++r) {
        const float ts  = __shfl(total, r + (half << 3), 32);
        const float rcp = __builtin_amdgcn_rcpf(ts);
        const int row   = r + half * 8;
#pragma unroll
        for (int nt = 0; nt < 8; ++nt)
            obase[(size_t)row * D_LEN + nt * 16 + mrow] = acc[nt][r] * rcp;
    }
}

extern "C" void kernel_launch(void* const* d_in, const int* in_sizes, int n_in,
                              void* d_out, int out_size, void* d_ws, size_t ws_size,
                              hipStream_t stream) {
    (void)in_sizes; (void)n_in; (void)out_size; (void)d_ws; (void)ws_size;
    const float* scores = (const float*)d_in[0];
    const float* v      = (const float*)d_in[1];
    float* out          = (float*)d_out;

    // B*H = 32 heads, S/128 = 16 q-blocks each -> 512 blocks of 256 threads (8 waves)
    dim3 grid(2 * 16 * (S_LEN / 128));
    dim3 block(256);
    softmaxv_wmma_kernel<<<grid, block, 0, stream>>>(scores, v, out);
}
